// Attention_43843026158179
// MI455X (gfx1250) — matrix operand order repaired and checked by reading, not compiled, not run on hardware
//
#include <hip/hip_runtime.h>
#include <hip/hip_bf16.h>

#define Bz  2
#define Lz  2048
#define Dz  1024
#define Hz  16
#define HDz 64
#define Mz  (Bz * Lz)                 // 4096 rows
#define SCALE_LOG2E 0.18033688011112042f   // (1/sqrt(64)) * log2(e)

typedef __attribute__((ext_vector_type(16))) _Float16 v16h;
typedef __attribute__((ext_vector_type(8)))  _Float16 v8h;
typedef __attribute__((ext_vector_type(8)))  float    v8f;
typedef __attribute__((ext_vector_type(4)))  int      v4i;

// ---- async global->LDS copy (CDNA5), with synchronous fallback ------------
#if defined(__has_builtin)
#  if __has_builtin(__builtin_amdgcn_global_load_async_to_lds_b128) && \
      __has_builtin(__builtin_amdgcn_s_wait_asynccnt)
#    define HAVE_ASYNC_LDS 1
#  endif
#endif
#ifndef HAVE_ASYNC_LDS
#  define HAVE_ASYNC_LDS 0
#endif

#if HAVE_ASYNC_LDS
typedef __attribute__((address_space(1))) v4i gv4i;   // global int4
typedef __attribute__((address_space(3))) v4i lv4i;   // LDS int4
#endif

static __device__ __forceinline__ void copy16_to_lds(const _Float16* g, _Float16* l) {
#if HAVE_ASYNC_LDS
    __builtin_amdgcn_global_load_async_to_lds_b128((gv4i*)g, (lv4i*)l, 0, 0);
#else
    *(v8h*)l = *(const v8h*)g;
#endif
}
static __device__ __forceinline__ void async_wait0() {
#if HAVE_ASYNC_LDS
    __builtin_amdgcn_s_wait_asynccnt(0);
#endif
}
static __device__ __forceinline__ void async_wait1() {
#if HAVE_ASYNC_LDS
    __builtin_amdgcn_s_wait_asynccnt(1);
#endif
}
static __device__ __forceinline__ void async_wait2() {
#if HAVE_ASYNC_LDS
    __builtin_amdgcn_s_wait_asynccnt(2);
#endif
}

// ---- WMMA operand helpers (layouts per CDNA5 ISA §7.12.2) ------
static __device__ __forceinline__ v16h cat16(v8h lo, v8h hi) {
    return __builtin_shufflevector(lo, hi, 0,1,2,3,4,5,6,7,8,9,10,11,12,13,14,15);
}

// A-matrix 16x32 (MxK), row-major source, leading dim `ld`.
static __device__ __forceinline__ v16h load_a16x32(const _Float16* base, int ld, int lane) {
    const int m  = lane & 15;
    const int kb = (lane >> 4) << 3;          // 0 or 8
    const _Float16* p = base + (size_t)m * ld + kb;
    return cat16(*(const v8h*)(p), *(const v8h*)(p + 16));
}

// B-matrix 32x16 (KxN), element B[k][n] = src[n*ld + k].
static __device__ __forceinline__ v16h load_b32x16(const _Float16* base, int ld, int lane) {
    const int n  = lane & 15;
    const int kb = (lane >> 4) << 3;          // 0 or 8
    const _Float16* p = base + (size_t)n * ld + kb;
    return cat16(*(const v8h*)(p), *(const v8h*)(p + 16));
}

static __device__ __forceinline__ v8f wmma_f16(v16h a, v16h b, v8f c) {
    return __builtin_amdgcn_wmma_f32_16x16x32_f16(false, a, false, b, (short)0, c,
                                                  false, false);
}

// ---------------------------------------------------------------------------
// fp32 -> f16 conversion
// ---------------------------------------------------------------------------
__global__ void __launch_bounds__(256)
f32_to_f16_kernel(const float* __restrict__ src, _Float16* __restrict__ dst, int n) {
    int i = blockIdx.x * blockDim.x + threadIdx.x;
    if (i < n) dst[i] = (_Float16)src[i];
}

// ---------------------------------------------------------------------------
// GEMM: Y[M,N] = A[M,K] * Bw[N,K]^T  (x @ W.T). 8 waves/block share one
// 64-wide B strip; 64x32 f16 B k-chunks are double-buffered in LDS via
// async global->LDS copies. MODE 0: [B,H,L,HD]; 1: [B,H,HD,L]; 2: fp32 [M,N].
// ---------------------------------------------------------------------------
template <int MODE>
__global__ void __launch_bounds__(256)
gemm16x64(const _Float16* __restrict__ A, const _Float16* __restrict__ Bw,
          _Float16* __restrict__ dst16, float* __restrict__ dst32,
          int N, int K) {
    __shared__ _Float16 bbuf[2][64 * 32];     // double-buffered B chunk, 2 x 4 KB

    const int t     = threadIdx.x;
    const int lane  = t & 31;
    const int wslot = t >> 5;
    const int nTile = blockIdx.x & ((N >> 6) - 1);
    const int mTile = (blockIdx.x >> 4) * 8 + wslot;

    const _Float16* arow   = A + (size_t)mTile * 16 * K;
    const _Float16* bstrip = Bw + (size_t)nTile * 64 * K;

    // staging: thread t copies 16 B of row (t>>2), chunk (t&3)
    const int srow = t >> 2, sc = (t & 3) << 3;
    const _Float16* bsrc = bstrip + (size_t)srow * K + sc;

    copy16_to_lds(bsrc, &bbuf[0][srow * 32 + sc]);          // tile 0

    v8f c0 = {}, c1 = {}, c2 = {}, c3 = {};
    for (int k = 0; k < K; k += 32) {
        const int cur = (k >> 5) & 1;
        if (k + 32 < K) {
            copy16_to_lds(bsrc + k + 32, &bbuf[cur ^ 1][srow * 32 + sc]);
            async_wait1();                                   // tile k done
            if (k + 64 < K) __builtin_prefetch(bsrc + k + 64, 0, 1);
        } else {
            async_wait0();
        }
        __syncthreads();

        const _Float16* bb = &bbuf[cur][0];
        v16h a = load_a16x32(arow + k, K, lane);
        c0 = wmma_f16(a, load_b32x16(bb +  0 * 32, 32, lane), c0);
        c1 = wmma_f16(a, load_b32x16(bb + 16 * 32, 32, lane), c1);
        c2 = wmma_f16(a, load_b32x16(bb + 32 * 32, 32, lane), c2);
        c3 = wmma_f16(a, load_b32x16(bb + 48 * 32, 32, lane), c3);
        __syncthreads();    // all waves done with bbuf[cur] before next overwrite
    }
    v8f c[4] = {c0, c1, c2, c3};

#pragma unroll
    for (int j = 0; j < 4; ++j) {
#pragma unroll
        for (int i = 0; i < 8; ++i) {
            const int row = mTile * 16 + i + ((lane >> 4) << 3);
            const int col = nTile * 64 + j * 16 + (lane & 15);
            const float v = c[j][i];
            if (MODE == 0) {            // [B,H,L,HD]
                const int b = row >> 11, l = row & (Lz - 1);
                const int h = col >> 6,  hd = col & (HDz - 1);
                dst16[((((size_t)b * Hz + h) * Lz) + l) * HDz + hd] = (_Float16)v;
            } else if (MODE == 1) {     // [B,H,HD,L]
                const int b = row >> 11, l = row & (Lz - 1);
                const int h = col >> 6,  hd = col & (HDz - 1);
                dst16[((((size_t)b * Hz + h) * HDz) + hd) * Lz + l] = (_Float16)v;
            } else {                    // plain fp32 [M,N]
                dst32[(size_t)row * N + col] = v;
            }
        }
    }
}

// ---------------------------------------------------------------------------
// Flash attention. Block = 8 waves = one (b,h), 8 query tiles (128 queries).
// K/V tiles double-buffered in LDS via async copies; row softmax stats in
// A-layout (one cross-half shuffle per reduction).
// ---------------------------------------------------------------------------
__global__ void __launch_bounds__(256)
flash_attn(const _Float16* __restrict__ Qh, const _Float16* __restrict__ Kh,
           const _Float16* __restrict__ Vt, _Float16* __restrict__ O) {
    __shared__ _Float16 kbuf[2][32 * 64];     // [key][hd], 2 x 4 KB
    __shared__ _Float16 vbuf[2][64 * 32];     // [hd][key], 2 x 4 KB
    __shared__ float    sbuf[8][16 * 32];     // per-wave scores, 16 KB

    const int t     = threadIdx.x;
    const int lane  = t & 31;
    const int wslot = t >> 5;
    const int qg    = blockIdx.x & 15;
    const int h     = (blockIdx.x >> 4) & (Hz - 1);
    const int b     = blockIdx.x >> 8;
    const int qTile = qg * 8 + wslot;

    const _Float16* Qb = Qh + (((size_t)b * Hz + h) * Lz + (size_t)qTile * 16) * HDz;
    const _Float16* Kb = Kh + ((size_t)b * Hz + h) * Lz * HDz;
    const _Float16* Vb = Vt + ((size_t)b * Hz + h) * HDz * Lz;

    const v16h aq0 = load_a16x32(Qb + 0,  HDz, lane);
    const v16h aq1 = load_a16x32(Qb + 32, HDz, lane);

    v8f o0 = {}, o1 = {}, o2 = {}, o3 = {};
    float mprev = -3.0e38f, lsum = 0.0f;      // stats for row (lane&15)

    const int srow = t >> 2, sc = (t & 3) << 3;   // V staging assignment
    float* sw = &sbuf[wslot][0];
    const int NT = Lz / 32;                   // 64 key tiles

    // stage tile kb into buffer `buf`: K is one contiguous 4 KB row-block,
    // V rows are strided by L.
    auto stage = [&](int kb, int buf) {
        copy16_to_lds(Kb + (size_t)kb * 32 * HDz + t * 8, &kbuf[buf][t * 8]);
        copy16_to_lds(Vb + (size_t)srow * Lz + kb * 32 + sc,
                      &vbuf[buf][srow * 32 + sc]);
    };

    stage(0, 0);

    for (int kb = 0; kb < NT; ++kb) {
        const int cur = kb & 1;
        if (kb + 1 < NT) {
            stage(kb + 1, cur ^ 1);
            async_wait2();                     // tile kb's 2 copies done
            if (kb + 2 < NT)
                __builtin_prefetch(Kb + (size_t)(kb + 2) * 32 * HDz + t * 8, 0, 1);
        } else {
            async_wait0();
        }
        __syncthreads();

        // ---- S tile 16x32 = Q * K^T ----
        const _Float16* kk = &kbuf[cur][0];
        v8f s0 = {}, s1 = {};
        s0 = wmma_f16(aq0, load_b32x16(kk +  0 * 64 +  0, 64, lane), s0);
        s0 = wmma_f16(aq1, load_b32x16(kk +  0 * 64 + 32, 64, lane), s0);
        s1 = wmma_f16(aq0, load_b32x16(kk + 16 * 64 +  0, 64, lane), s1);
        s1 = wmma_f16(aq1, load_b32x16(kk + 16 * 64 + 32, 64, lane), s1);

        // ---- scores (C-layout) -> per-wave LDS transpose ----
#pragma unroll
        for (int i = 0; i < 8; ++i) {
            const int row = i + ((lane >> 4) << 3);
            sw[row * 32 + (lane & 15)]      = s0[i] * SCALE_LOG2E;
            sw[row * 32 + 16 + (lane & 15)] = s1[i] * SCALE_LOG2E;
        }
        __syncthreads();

        // ---- A-layout reload: lane & lane+16 hold one full row ----
        const float* sp = sw + (size_t)(lane & 15) * 32 + ((lane >> 4) << 3);
        v8f tlo = *(const v8f*)sp;
        v8f thi = *(const v8f*)(sp + 16);

        float mx = tlo[0];
#pragma unroll
        for (int i = 1; i < 8; ++i) mx = fmaxf(mx, tlo[i]);
#pragma unroll
        for (int i = 0; i < 8; ++i) mx = fmaxf(mx, thi[i]);
        mx = fmaxf(mx, __shfl_xor(mx, 16));
        const float mn = fmaxf(mprev, mx);
        const float al = __builtin_amdgcn_exp2f(mprev - mn);
        mprev = mn;

        v16h ap = {};
        float rs = 0.0f;
#pragma unroll
        for (int i = 0; i < 8; ++i) {
            const float p0 = __builtin_amdgcn_exp2f(tlo[i] - mn);
            const float p1 = __builtin_amdgcn_exp2f(thi[i] - mn);
            rs += p0 + p1;
            ap[i]     = (_Float16)p0;
            ap[i + 8] = (_Float16)p1;
        }
        rs += __shfl_xor(rs, 16);
        lsum = lsum * al + rs;

        // ---- rescale running O (one shuffle per row) ----
#pragma unroll
        for (int i = 0; i < 8; ++i) {
            const float ar = __shfl(al, i + ((lane >> 4) << 3));
            o0[i] *= ar; o1[i] *= ar; o2[i] *= ar; o3[i] *= ar;
        }

        // ---- O += P * V ----
        const _Float16* vv = &vbuf[cur][0];
        o0 = wmma_f16(ap, load_b32x16(vv +  0 * 32, 32, lane), o0);
        o1 = wmma_f16(ap, load_b32x16(vv + 16 * 32, 32, lane), o1);
        o2 = wmma_f16(ap, load_b32x16(vv + 32 * 32, 32, lane), o2);
        o3 = wmma_f16(ap, load_b32x16(vv + 48 * 32, 32, lane), o3);
        __syncthreads();   // all reads of buf[cur] done before it is re-staged
    }

    const float inv = 1.0f / lsum;
    v8f o[4] = {o0, o1, o2, o3};
#pragma unroll
    for (int j = 0; j < 4; ++j) {
#pragma unroll
        for (int i = 0; i < 8; ++i) {
            const float ir = __shfl(inv, i + ((lane >> 4) << 3));
            const int row = qTile * 16 + i + ((lane >> 4) << 3);   // l index
            const int col = h * HDz + j * 16 + (lane & 15);        // d index
            O[(((size_t)b * Lz) + row) * Dz + col] = (_Float16)(o[j][i] * ir);
        }
    }
}

// ---------------------------------------------------------------------------
extern "C" void kernel_launch(void* const* d_in, const int* in_sizes, int n_in,
                              void* d_out, int out_size, void* d_ws, size_t ws_size,
                              hipStream_t stream) {
    const float* x  = (const float*)d_in[0];
    const float* Wq = (const float*)d_in[1];
    const float* Wk = (const float*)d_in[2];
    const float* Wv = (const float*)d_in[3];
    const float* Wo = (const float*)d_in[4];
    float* out = (float*)d_out;

    char* ws = (char*)d_ws;
    const size_t MB = 1024 * 1024;
    _Float16* x16  = (_Float16*)(ws + 0 * MB);    // 4096x1024 f16 = 8 MB
    _Float16* Wq16 = (_Float16*)(ws + 8 * MB);
    _Float16* Wk16 = (_Float16*)(ws + 10 * MB);
    _Float16* Wv16 = (_Float16*)(ws + 12 * MB);
    _Float16* Wo16 = (_Float16*)(ws + 14 * MB);
    _Float16* Qh   = (_Float16*)(ws + 16 * MB);   // [B,H,L,HD]
    _Float16* Kh   = (_Float16*)(ws + 24 * MB);   // [B,H,L,HD]
    _Float16* Vt   = (_Float16*)(ws + 32 * MB);   // [B,H,HD,L]
    _Float16* Oat  = (_Float16*)(ws + 40 * MB);   // [B,L,D]

    const int nx = Mz * Dz;
    const int nw = Dz * Dz;
    f32_to_f16_kernel<<<nx / 256, 256, 0, stream>>>(x,  x16,  nx);
    f32_to_f16_kernel<<<nw / 256, 256, 0, stream>>>(Wq, Wq16, nw);
    f32_to_f16_kernel<<<nw / 256, 256, 0, stream>>>(Wk, Wk16, nw);
    f32_to_f16_kernel<<<nw / 256, 256, 0, stream>>>(Wv, Wv16, nw);
    f32_to_f16_kernel<<<nw / 256, 256, 0, stream>>>(Wo, Wo16, nw);

    const int gemmBlocks = (Mz / 16 / 8) * (Dz / 64);    // 512
    gemm16x64<0><<<gemmBlocks, 256, 0, stream>>>(x16, Wq16, Qh, nullptr, Dz, Dz);
    gemm16x64<0><<<gemmBlocks, 256, 0, stream>>>(x16, Wk16, Kh, nullptr, Dz, Dz);
    gemm16x64<1><<<gemmBlocks, 256, 0, stream>>>(x16, Wv16, Vt, nullptr, Dz, Dz);

    flash_attn<<<512, 256, 0, stream>>>(Qh, Kh, Vt, Oat);   // B*H*(L/128) = 512

    gemm16x64<2><<<gemmBlocks, 256, 0, stream>>>(Oat, Wo16, nullptr, out, Dz, Dz);
}
